// TokenRoutedMLPTriton_76209899700397
// MI455X (gfx1250) — matrix operand order repaired and checked by reading, not compiled, not run on hardware
//
#include <hip/hip_runtime.h>

namespace {
constexpr int kB = 8, kS = 4096, kH = 1024, kE = 4, kEI = 1024;
constexpr int kT = kB * kS;          // 32768 tokens
constexpr int kVocab = 100000;
constexpr int kTM = 32;              // tokens per workgroup tile
constexpr int kXStr = 1032;          // padded LDS row stride (halves)
constexpr size_t kWElems = (size_t)kE * kH * kEI;   // elements per weight matrix
}

typedef __attribute__((ext_vector_type(16))) __bf16 v16bf;
typedef __attribute__((ext_vector_type(8)))  float  v8f;

union BF16x16 { v16bf v; unsigned short u[16]; uint4 q[2]; };

__device__ __forceinline__ unsigned short f2bf(float f) {
  union { float f; unsigned u; } x; x.f = f;
  unsigned r = x.u + 0x7FFFu + ((x.u >> 16) & 1u);   // round-to-nearest-even
  return (unsigned short)(r >> 16);
}

// ---- CDNA5 async global->LDS copy (ASYNCcnt-tracked) ----
__device__ __forceinline__ unsigned lds_off(const void* shared_ptr) {
  // Flat shared pointers: addr[31:0] is the LDS byte offset (aperture rule).
  return (unsigned)(unsigned long long)(uintptr_t)shared_ptr;
}
__device__ __forceinline__ void async_copy_b128(unsigned dstLdsOff, const void* src) {
  asm volatile("global_load_async_to_lds_b128 %0, %1, off"
               :: "v"(dstLdsOff), "v"(src) : "memory");
}
__device__ __forceinline__ void wait_async0() {
  asm volatile("s_wait_asynccnt 0x0" ::: "memory");
}

// A fragment (16x32 bf16, M x K) from an LDS tile, row-major stride kXStr.
__device__ __forceinline__ v16bf load_a_lds(const unsigned short* tile, int rowBase,
                                            int k, int lane) {
  const int row = rowBase + (lane & 15);
  const int h   = (lane >> 4) * 8;
  const unsigned short* p = tile + row * kXStr + k + h;
  BF16x16 r;
  r.q[0] = *reinterpret_cast<const uint4*>(p);        // K = k+h .. k+h+7
  r.q[1] = *reinterpret_cast<const uint4*>(p + 16);   // K = k+h+16 .. k+h+23
  return r.v;
}

// B fragment (32x16 bf16, K x N): lane l holds column N=l%16, K = 8h+s / 16+8h+s (table order).
__device__ __forceinline__ v16bf load_b_bf16(const unsigned short* p, size_t ld) {
  BF16x16 r;
#pragma unroll
  for (int s = 0; s < 8; ++s) { r.u[s] = p[s * ld]; r.u[8 + s] = p[(16 + s) * ld]; }
  return r.v;
}

// Fallback: B fragment from f32 weights with on-the-fly conversion.
__device__ __forceinline__ v16bf load_b_f32(const float* p, size_t ld) {
  BF16x16 r;
#pragma unroll
  for (int s = 0; s < 8; ++s) { r.u[s] = f2bf(p[s * ld]); r.u[8 + s] = f2bf(p[(16 + s) * ld]); }
  return r.v;
}

__device__ __forceinline__ v8f wmma_bf16(v16bf a, v16bf b, v8f c) {
  return __builtin_amdgcn_wmma_f32_16x16x32_bf16(false, a, false, b, (short)0, c,
                                                 false, false);
}

// ---------------- workspace layout ----------------
// ints: [0..3] counts  [4..7] cursors  [8..11] token offsets
//       [12..16] tile offsets ([16]=total tiles)
//       [32 .. 32+kT) expert_ids   [32+kT .. 32+2*kT) token lists
// bytes kWbOff..: bf16 weights, gate | up | down (kWElems halves each)

__global__ void init_kernel(int* wsI) {
  if (threadIdx.x < 32) wsI[threadIdx.x] = 0;
}

__global__ __launch_bounds__(256) void route_kernel(const int* __restrict__ token_ids,
                                                    const float* __restrict__ mu,
                                                    const float* __restrict__ rw,
                                                    int* wsI) {
  __shared__ float Wl[kE * kH];                       // 16 KB router weights
  const int tid = threadIdx.x;
  for (int i = tid; i < kE * kH; i += 256) Wl[i] = rw[i];
  __syncthreads();

  const int lane = tid & 31;
  const int t = blockIdx.x * 8 + (tid >> 5);          // one wave32 per token
  const float* m = mu + (size_t)t * kH;
  float s0 = 0.f, s1 = 0.f, s2 = 0.f, s3 = 0.f;
  for (int h = lane; h < kH; h += 32) {
    const float mv = m[h];
    s0 += mv * Wl[h];
    s1 += mv * Wl[kH + h];
    s2 += mv * Wl[2 * kH + h];
    s3 += mv * Wl[3 * kH + h];
  }
  for (int off = 16; off > 0; off >>= 1) {
    s0 += __shfl_xor(s0, off, 32);
    s1 += __shfl_xor(s1, off, 32);
    s2 += __shfl_xor(s2, off, 32);
    s3 += __shfl_xor(s3, off, 32);
  }
  if (lane == 0) {
    int tc = token_ids[t];
    tc = tc < 0 ? 0 : (tc > kVocab - 1 ? kVocab - 1 : tc);
    const int be = tc & (kE - 1);                     // tid % 4
    const float L[4] = {s0, s1, s2, s3};
    float best = -3.4e38f; int bi = 0;
    for (int e = 0; e < kE; ++e) {
      const float v = L[e] + (e == be ? 10.f : 0.f);
      if (v > best) { best = v; bi = e; }             // first-max tie break
    }
    (wsI + 32)[t] = bi;
    atomicAdd(&wsI[bi], 1);
  }
}

__global__ void scan_kernel(int* wsI) {
  if (threadIdx.x == 0) {
    int acc = 0, tiles = 0;
    for (int e = 0; e < kE; ++e) {
      wsI[8 + e]  = acc;
      wsI[12 + e] = tiles;
      acc   += wsI[e];
      tiles += (wsI[e] + kTM - 1) / kTM;
    }
    wsI[16] = tiles;
  }
}

__global__ __launch_bounds__(256) void build_lists_kernel(int* wsI) {
  const int t = blockIdx.x * 256 + threadIdx.x;
  const int e = (wsI + 32)[t];
  const int pos = atomicAdd(&wsI[4 + e], 1);
  (wsI + 32 + kT)[wsI[8 + e] + pos] = t;
}

// One-time f32 -> bf16 weight conversion into workspace (gate|up|down).
__global__ __launch_bounds__(256) void convert_weights_kernel(
    const float* __restrict__ g, const float* __restrict__ u,
    const float* __restrict__ d, unsigned short* __restrict__ wb) {
  const size_t i = ((size_t)blockIdx.x * 256 + threadIdx.x) * 8;
  const float* src;
  if (i < kWElems)            src = g + i;
  else if (i < 2 * kWElems)   src = u + (i - kWElems);
  else                        src = d + (i - 2 * kWElems);
  const float4 f0 = *reinterpret_cast<const float4*>(src);
  const float4 f1 = *reinterpret_cast<const float4*>(src + 4);
  uint4 o;
  o.x = (unsigned)f2bf(f0.x) | ((unsigned)f2bf(f0.y) << 16);
  o.y = (unsigned)f2bf(f0.z) | ((unsigned)f2bf(f0.w) << 16);
  o.z = (unsigned)f2bf(f1.x) | ((unsigned)f2bf(f1.y) << 16);
  o.w = (unsigned)f2bf(f1.z) | ((unsigned)f2bf(f1.w) << 16);
  *reinterpret_cast<uint4*>(wb + i) = o;
}

template <bool PRE>
__global__ __launch_bounds__(256) void moe_gemm_kernel(
    const float* __restrict__ x,
    const float* __restrict__ gate, const float* __restrict__ up,
    const float* __restrict__ down,
    const unsigned short* __restrict__ wbf,
    const int* __restrict__ wsI, float* __restrict__ out) {
  __shared__ __align__(16) unsigned short Xs[kTM * kXStr];  // 66 KB bf16 X tile
  __shared__ __align__(16) unsigned short Ps[kTM * kXStr];  // 66 KB bf16 intermediate
  __shared__ __align__(16) float Xstage[kTM * 128];         // 16 KB async staging

  const int total = wsI[16];
  const int bid = blockIdx.x;
  if (bid >= total) return;
  int e = 0;
  for (int i = 1; i < kE; ++i) if (bid >= wsI[12 + i]) e = i;
  const int tileIdx = bid - wsI[12 + e];
  const int cnt = wsI[e];
  const int tokenStart = wsI[8 + e] + tileIdx * kTM;
  const int nTok = min(kTM, cnt - tileIdx * kTM);
  const int* lists = wsI + 32 + kT;

  const float* gp32 = gate + (size_t)e * kH * kEI;
  const float* up32 = up   + (size_t)e * kH * kEI;
  const float* dn32 = down + (size_t)e * kEI * kH;
  const unsigned short* gp16 = wbf + (size_t)e * kH * kEI;
  const unsigned short* up16 = wbf + kWElems + (size_t)e * kH * kEI;
  const unsigned short* dn16 = wbf + 2 * kWElems + (size_t)e * kEI * kH;

  const int tid = threadIdx.x;

  // ---- stage gathered X rows: async f32 -> LDS, then convert to bf16 tile ----
  // 8 column-chunks of 128; per chunk 1024 x 16B transfers (4 per thread).
  for (int cc = 0; cc < 8; ++cc) {
    int idx = tid;
#pragma unroll
    for (int j = 0; j < 4; ++j, idx += 256) {
      const int row = idx >> 5;                 // 0..31
      const int c4  = (idx & 31) * 4;           // 0..124
      const int rc  = row < nTok ? row : nTok - 1;   // clamp padded rows
      const float* src = x + (size_t)lists[tokenStart + rc] * kH + cc * 128 + c4;
      async_copy_b128(lds_off(&Xstage[row * 128 + c4]), src);
    }
    wait_async0();
    __syncthreads();
    idx = tid;
#pragma unroll
    for (int j = 0; j < 4; ++j, idx += 256) {
      const int row = idx >> 5;
      const int c4  = (idx & 31) * 4;
      const float4 v = *reinterpret_cast<const float4*>(&Xstage[row * 128 + c4]);
      uint2 hv;
      hv.x = (unsigned)f2bf(v.x) | ((unsigned)f2bf(v.y) << 16);
      hv.y = (unsigned)f2bf(v.z) | ((unsigned)f2bf(v.w) << 16);
      *reinterpret_cast<uint2*>(Xs + row * kXStr + cc * 128 + c4) = hv;
    }
    __syncthreads();
  }

  const int lane = tid & 31;
  const int wv   = tid >> 5;
  const int hi   = lane >> 4;
  const int ln   = lane & 15;

  // ---- GEMM1: [32,1024] x (gate|up) + fused SwiGLU -> Ps ----
  for (int colN = wv * 16; colN < kEI; colN += 128) {
    v8f g0 = {}, g1 = {}, u0 = {}, u1 = {};
    if (PRE) {
      const unsigned short* bg = gp16 + (size_t)(8 * hi) * kEI + colN + ln;
      const unsigned short* bu = up16 + (size_t)(8 * hi) * kEI + colN + ln;
      for (int k = 0; k < kH; k += 32) {
        const v16bf a0 = load_a_lds(Xs, 0, k, lane);
        const v16bf a1 = load_a_lds(Xs, 16, k, lane);
        const v16bf Bg = load_b_bf16(bg, kEI);
        const v16bf Bu = load_b_bf16(bu, kEI);
        bg += (size_t)32 * kEI;
        bu += (size_t)32 * kEI;
        if (k + 32 < kH) { __builtin_prefetch(bg, 0, 1); __builtin_prefetch(bu, 0, 1); }
        g0 = wmma_bf16(a0, Bg, g0);
        g1 = wmma_bf16(a1, Bg, g1);
        u0 = wmma_bf16(a0, Bu, u0);
        u1 = wmma_bf16(a1, Bu, u1);
      }
    } else {
      const float* bg = gp32 + (size_t)(8 * hi) * kEI + colN + ln;
      const float* bu = up32 + (size_t)(8 * hi) * kEI + colN + ln;
      for (int k = 0; k < kH; k += 32) {
        const v16bf a0 = load_a_lds(Xs, 0, k, lane);
        const v16bf a1 = load_a_lds(Xs, 16, k, lane);
        const v16bf Bg = load_b_f32(bg, kEI);
        const v16bf Bu = load_b_f32(bu, kEI);
        bg += (size_t)32 * kEI;
        bu += (size_t)32 * kEI;
        if (k + 32 < kH) { __builtin_prefetch(bg, 0, 1); __builtin_prefetch(bu, 0, 1); }
        g0 = wmma_bf16(a0, Bg, g0);
        g1 = wmma_bf16(a1, Bg, g1);
        u0 = wmma_bf16(a0, Bu, u0);
        u1 = wmma_bf16(a1, Bu, u1);
      }
    }
#pragma unroll
    for (int r = 0; r < 8; ++r) {
      {
        const float gv = g0[r], uv = u0[r];
        const float pv = gv / (1.f + __expf(-gv)) * uv;
        Ps[(r + 8 * hi) * kXStr + colN + ln] = f2bf(pv);
      }
      {
        const float gv = g1[r], uv = u1[r];
        const float pv = gv / (1.f + __expf(-gv)) * uv;
        Ps[(16 + r + 8 * hi) * kXStr + colN + ln] = f2bf(pv);
      }
    }
  }
  __syncthreads();

  // ---- GEMM2: out = Ps @ down ----
  for (int colN = wv * 16; colN < kH; colN += 128) {
    v8f o0 = {}, o1 = {};
    if (PRE) {
      const unsigned short* bd = dn16 + (size_t)(8 * hi) * kH + colN + ln;
      for (int k = 0; k < kEI; k += 32) {
        const v16bf a0 = load_a_lds(Ps, 0, k, lane);
        const v16bf a1 = load_a_lds(Ps, 16, k, lane);
        const v16bf Bd = load_b_bf16(bd, kH);
        bd += (size_t)32 * kH;
        if (k + 32 < kEI) __builtin_prefetch(bd, 0, 1);
        o0 = wmma_bf16(a0, Bd, o0);
        o1 = wmma_bf16(a1, Bd, o1);
      }
    } else {
      const float* bd = dn32 + (size_t)(8 * hi) * kH + colN + ln;
      for (int k = 0; k < kEI; k += 32) {
        const v16bf a0 = load_a_lds(Ps, 0, k, lane);
        const v16bf a1 = load_a_lds(Ps, 16, k, lane);
        const v16bf Bd = load_b_f32(bd, kH);
        bd += (size_t)32 * kH;
        if (k + 32 < kEI) __builtin_prefetch(bd, 0, 1);
        o0 = wmma_bf16(a0, Bd, o0);
        o1 = wmma_bf16(a1, Bd, o1);
      }
    }
#pragma unroll
    for (int r = 0; r < 8; ++r) {
      const int r0 = r + 8 * hi;
      if (r0 < nTok) out[(size_t)lists[tokenStart + r0] * kH + colN + ln] = o0[r];
      const int r1 = 16 + r + 8 * hi;
      if (r1 < nTok) out[(size_t)lists[tokenStart + r1] * kH + colN + ln] = o1[r];
    }
  }
}

extern "C" void kernel_launch(void* const* d_in, const int* in_sizes, int n_in,
                              void* d_out, int out_size, void* d_ws, size_t ws_size,
                              hipStream_t stream) {
  (void)in_sizes; (void)n_in; (void)out_size;
  const float* x   = (const float*)d_in[0];  // hidden_states [T,H]
  const int*   tok = (const int*)d_in[1];    // token_ids [T]
  const float* mu  = (const float*)d_in[2];  // mu [T,H]
  const float* gp  = (const float*)d_in[3];  // gate_proj [E,H,EI]
  const float* up  = (const float*)d_in[4];  // up_proj   [E,H,EI]
  const float* dn  = (const float*)d_in[5];  // down_proj [E,EI,H]
  const float* rw  = (const float*)d_in[6];  // mu_router_w [E,H]
  float* out = (float*)d_out;
  int* wsI = (int*)d_ws;

  constexpr size_t kIntBytes = (size_t)(32 + 2 * kT) * sizeof(int);
  constexpr size_t kWbOff = ((kIntBytes + 4095) / 4096) * 4096;
  const size_t need = kWbOff + 3 * kWElems * sizeof(unsigned short);
  unsigned short* wbf = (unsigned short*)((char*)d_ws + kWbOff);
  const bool pre = ws_size >= need;

  init_kernel<<<1, 64, 0, stream>>>(wsI);
  route_kernel<<<kT / 8, 256, 0, stream>>>(tok, mu, rw, wsI);
  scan_kernel<<<1, 1, 0, stream>>>(wsI);
  build_lists_kernel<<<kT / 256, 256, 0, stream>>>(wsI);
  const int gemmGrid = kT / kTM + kE;
  if (pre) {
    convert_weights_kernel<<<(int)(3 * kWElems / 2048), 256, 0, stream>>>(gp, up, dn, wbf);
    moe_gemm_kernel<true><<<gemmGrid, 256, 0, stream>>>(x, gp, up, dn, wbf, wsI, out);
  } else {
    moe_gemm_kernel<false><<<gemmGrid, 256, 0, stream>>>(x, gp, up, dn, wbf, wsI, out);
  }
}
